// Attn_57011395887454
// MI455X (gfx1250) — compile-verified, matrix operand order checked by reading, not run on hardware
//
#include <hip/hip_runtime.h>

typedef __bf16 bf16;
typedef __attribute__((ext_vector_type(16))) __bf16 v16bf;
typedef __attribute__((ext_vector_type(8)))  __bf16 v8bf;
typedef __attribute__((ext_vector_type(8)))  float  v8f;

constexpr int BB = 4, TT = 2048, DD = 1024, NH = 16, DH = 64;
constexpr int D3 = 3 * DD;
constexpr float SCALE = 0.125f;  // 1/sqrt(64)

// ---------- helpers ----------

__device__ inline bf16 f2bf(float f) {
  unsigned u = __builtin_bit_cast(unsigned, f);
  u += 0x7FFFu + ((u >> 16) & 1u);           // round-to-nearest-even
  unsigned short h = (unsigned short)(u >> 16);
  return __builtin_bit_cast(bf16, h);
}

// Load a 16-element bf16 WMMA fragment for this lane.
// p32 points at the start of the 32-wide K-chunk of this lane's row/column.
// Per ISA 16-bit A/B layout: lane half 0 holds K={0..7,16..23},
// lane half 1 holds K={8..15,24..31} -> two aligned 16-byte loads.
__device__ inline v16bf load_frag(const bf16* p32, int hi) {
  v8bf lo = *reinterpret_cast<const v8bf*>(p32 + 8 * hi);
  v8bf hv = *reinterpret_cast<const v8bf*>(p32 + 16 + 8 * hi);
  return __builtin_shufflevector(lo, hv, 0, 1, 2, 3, 4, 5, 6, 7,
                                         8, 9, 10, 11, 12, 13, 14, 15);
}

__device__ inline v8f wmma_bf16(v16bf a, v16bf b, v8f c) {
  // (neg_a, A, neg_b, B, c_mod, C, reuse_a, reuse_b)
  return __builtin_amdgcn_wmma_f32_16x16x32_bf16(false, a, false, b,
                                                 (short)0, c, false, false);
}

// ---------- elementwise kernels ----------

__global__ void f32_to_bf16(const float* __restrict__ in, bf16* __restrict__ out, long n) {
  long i = (long)blockIdx.x * blockDim.x + threadIdx.x;
  const long stride = (long)gridDim.x * blockDim.x;
  for (; i < n; i += stride) out[i] = f2bf(in[i]);
}

// vt[b,h,dh,t] = qkv[b,t, 2D + h*DH + dh]  (V transposed so P·V streams keys contiguously)
__global__ void transpose_v(const bf16* __restrict__ qkv, bf16* __restrict__ vt) {
  const long n = (long)BB * NH * DH * TT;
  long i = (long)blockIdx.x * blockDim.x + threadIdx.x;
  const long stride = (long)gridDim.x * blockDim.x;
  for (; i < n; i += stride) {
    const int t = (int)(i % TT);
    const int dh = (int)((i / TT) % DH);
    const int bh = (int)(i / ((long)TT * DH));
    const int batch = bh / NH, head = bh % NH;
    vt[i] = qkv[(size_t)(batch * TT + t) * D3 + 2 * DD + head * DH + dh];
  }
}

// ---------- C[m,n] = sum_k A[m,k] * W[n,k]  (x @ W^T), bf16 in, f32 accum ----------
// 256 threads = 8 waves; block tile 128x128; wave tile 64x32 (4x2 of 16x16)

template <bool STORE_BF16>
__global__ __launch_bounds__(256)
void gemm_xwT(const bf16* __restrict__ A, const bf16* __restrict__ W,
              const float* __restrict__ bias, float* __restrict__ outF,
              bf16* __restrict__ outB, int M, int N, int K) {
  const int lane = threadIdx.x & 31;
  const int wave = threadIdx.x >> 5;
  const int m = lane & 15, hi = lane >> 4;
  const int wm = wave & 1, wn = wave >> 1;
  const int m0 = blockIdx.y * 128 + wm * 64;
  const int n0 = blockIdx.x * 128 + wn * 32;

  v8f acc[4][2] = {};
  const bf16* arow[4];
  const bf16* wrow[2];
#pragma unroll
  for (int mi = 0; mi < 4; ++mi) arow[mi] = A + (size_t)(m0 + 16 * mi + m) * K;
#pragma unroll
  for (int ni = 0; ni < 2; ++ni) wrow[ni] = W + (size_t)(n0 + 16 * ni + m) * K;

  for (int kk = 0; kk < K; kk += 32) {
    v16bf af[4], bw[2];
#pragma unroll
    for (int mi = 0; mi < 4; ++mi) af[mi] = load_frag(arow[mi] + kk, hi);
#pragma unroll
    for (int ni = 0; ni < 2; ++ni) bw[ni] = load_frag(wrow[ni] + kk, hi);
    if (kk + 32 < K) {  // hint next K slab into cache (global_prefetch_b8)
      __builtin_prefetch(arow[0] + kk + 32, 0, 0);
      __builtin_prefetch(wrow[0] + kk + 32, 0, 0);
    }
#pragma unroll
    for (int mi = 0; mi < 4; ++mi)
#pragma unroll
      for (int ni = 0; ni < 2; ++ni)
        acc[mi][ni] = wmma_bf16(af[mi], bw[ni], acc[mi][ni]);
  }

#pragma unroll
  for (int mi = 0; mi < 4; ++mi)
#pragma unroll
    for (int ni = 0; ni < 2; ++ni) {
      const int col = n0 + 16 * ni + m;
#pragma unroll
      for (int r = 0; r < 8; ++r) {
        const int row = m0 + 16 * mi + r + 8 * hi;   // C layout: M = r + 8*hi
        if constexpr (STORE_BF16)
          outB[(size_t)row * N + col] = f2bf(acc[mi][ni][r]);
        else
          outF[(size_t)row * N + col] = acc[mi][ni][r] + bias[col];
      }
    }
}

// ---------- flash attention: one wave per 16-row Q tile per (b,h) ----------

__global__ __launch_bounds__(32)
void attn_fwd(const bf16* __restrict__ qkv, const bf16* __restrict__ vt,
              bf16* __restrict__ attn_out) {
  __shared__ bf16 p_lds[16 * 32];  // P tile re-layout buffer (C-frag -> A-frag)
  const int lane = threadIdx.x & 31;
  const int m = lane & 15, hi = lane >> 4;
  const int q0 = blockIdx.x * 16;
  const int bh = (int)blockIdx.y;
  const int batch = bh / NH, head = bh % NH;

  // Q fragments (16 rows x 64 dh = 2 K-chunks), loaded once
  const bf16* qrow = qkv + (size_t)(batch * TT + q0 + m) * D3 + head * DH;
  const v16bf qa0 = load_frag(qrow, hi);
  const v16bf qa1 = load_frag(qrow + 32, hi);

  v8f o[4] = {};              // 16 x 64 output accumulator (4 N-tiles)
  float m_run[8], l_run[8];   // per-row online softmax state (rows r + 8*hi)
#pragma unroll
  for (int r = 0; r < 8; ++r) { m_run[r] = -3.0e38f; l_run[r] = 0.f; }

  const int nkt = (q0 + 47) / 32;  // causal: keys 0 .. q0+15
  for (int kt = 0; kt < nkt; ++kt) {
    const int k0 = kt * 32;

    // S = Q @ K^T for 32 keys (two 16-key N-tiles, K-dim = dh = 64)
    v8f s[2];
#pragma unroll
    for (int nt = 0; nt < 2; ++nt) {
      const bf16* krow =
          qkv + (size_t)(batch * TT + k0 + nt * 16 + m) * D3 + DD + head * DH;
      v16bf b0 = load_frag(krow, hi);
      v16bf b1 = load_frag(krow + 32, hi);
      v8f z = {};
      z = wmma_bf16(qa0, b0, z);
      z = wmma_bf16(qa1, b1, z);
      s[nt] = z;
    }

    __syncthreads();  // previous iteration's p_lds reads done (S_NOP, 1 wave)

    float alpha[8];
#pragma unroll
    for (int r = 0; r < 8; ++r) {
      const int row = q0 + r + 8 * hi;
      float s0 = s[0][r] * SCALE;
      float s1 = s[1][r] * SCALE;
      if (k0 + m > row)      s0 = -3.0e38f;   // causal mask
      if (k0 + 16 + m > row) s1 = -3.0e38f;
      float mx = fmaxf(s0, s1);
#pragma unroll
      for (int d = 1; d < 16; d <<= 1) mx = fmaxf(mx, __shfl_xor(mx, d, 32));
      const float m_new = fmaxf(m_run[r], mx);
      alpha[r] = __expf(m_run[r] - m_new);
      const float p0 = __expf(s0 - m_new);
      const float p1 = __expf(s1 - m_new);
      float ps = p0 + p1;
#pragma unroll
      for (int d = 1; d < 16; d <<= 1) ps += __shfl_xor(ps, d, 32);
      l_run[r] = l_run[r] * alpha[r] + ps;
      m_run[r] = m_new;
      p_lds[(r + 8 * hi) * 32 + m] = f2bf(p0);
      p_lds[(r + 8 * hi) * 32 + 16 + m] = f2bf(p1);
    }
#pragma unroll
    for (int j = 0; j < 4; ++j)
#pragma unroll
      for (int r = 0; r < 8; ++r) o[j][r] *= alpha[r];

    __syncthreads();  // p_lds writes visible (DS in-order within wave)

    // O += P @ V  (contraction over 32 keys; V pre-transposed to [dh][t])
    const v16bf pa = load_frag(&p_lds[m * 32], hi);
#pragma unroll
    for (int j = 0; j < 4; ++j) {
      const bf16* vrow = vt + ((size_t)bh * DH + j * 16 + m) * TT + k0;
      v16bf bv = load_frag(vrow, hi);
      o[j] = wmma_bf16(pa, bv, o[j]);
    }
  }

#pragma unroll
  for (int r = 0; r < 8; ++r) l_run[r] = 1.0f / l_run[r];
#pragma unroll
  for (int j = 0; j < 4; ++j)
#pragma unroll
    for (int r = 0; r < 8; ++r) {
      const size_t row = (size_t)batch * TT + q0 + r + 8 * hi;
      attn_out[row * DD + head * DH + j * 16 + m] = f2bf(o[j][r] * l_run[r]);
    }
}

// ---------- launch ----------

extern "C" void kernel_launch(void* const* d_in, const int* in_sizes, int n_in,
                              void* d_out, int out_size, void* d_ws, size_t ws_size,
                              hipStream_t stream) {
  const float* x     = (const float*)d_in[0];
  const float* qkv_w = (const float*)d_in[1];
  const float* o_w   = (const float*)d_in[2];
  const float* o_b   = (const float*)d_in[3];
  float* out = (float*)d_out;

  char* ws = (char*)d_ws;
  size_t off = 0;
  auto take = [&](size_t elems) {
    bf16* p = (bf16*)(ws + off);
    off += ((elems * sizeof(bf16)) + 255) & ~(size_t)255;
    return p;
  };
  bf16* xb   = take((size_t)BB * TT * DD);        // x in bf16
  bf16* qwb  = take((size_t)3 * DD * DD);         // qkv_w in bf16
  bf16* owb  = take((size_t)DD * DD);             // o_w in bf16
  bf16* qkvb = take((size_t)BB * TT * 3 * DD);    // qkv activations
  bf16* vtb  = take((size_t)BB * NH * DH * TT);   // V transposed
  bf16* aob  = take((size_t)BB * TT * DD);        // attention output

  f32_to_bf16<<<2048, 256, 0, stream>>>(x,     xb,  (long)BB * TT * DD);
  f32_to_bf16<<<2048, 256, 0, stream>>>(qkv_w, qwb, (long)3 * DD * DD);
  f32_to_bf16<<<2048, 256, 0, stream>>>(o_w,   owb, (long)DD * DD);

  // qkv = x @ qkv_w^T : M=8192, N=3072, K=1024
  gemm_xwT<true><<<dim3((3 * DD) / 128, (BB * TT) / 128), 256, 0, stream>>>(
      xb, qwb, nullptr, nullptr, qkvb, BB * TT, 3 * DD, DD);

  transpose_v<<<2048, 256, 0, stream>>>(qkvb, vtb);

  attn_fwd<<<dim3(TT / 16, BB * NH), 32, 0, stream>>>(qkvb, vtb, aob);

  // out = attn @ o_w^T + o_b : M=8192, N=1024, K=1024 (fp32 out)
  gemm_xwT<false><<<dim3(DD / 128, (BB * TT) / 128), 256, 0, stream>>>(
      aob, owb, o_b, out, nullptr, BB * TT, DD, DD);
}
